// MultiHeadAttentionKVCache_60644938219779
// MI455X (gfx1250) — matrix operand order repaired and checked by reading, not compiled, not run on hardware
//
#include <hip/hip_runtime.h>
#include <hip/hip_bf16.h>

// ---------------------------------------------------------------------------
// MI455X (gfx1250) bf16 WMMA implementation of MHA + KV cache.
// All matrix math via v_wmma_f32_16x16x32_bf16 (fp32 accumulate).
// Flash attention: split unmasked/masked key loops; softmax in log2 domain
// (scale*log2e folded into Q); row-sums of P computed by an extra WMMA
// against a ones matrix (lands directly in the l-accumulator layout).
// ---------------------------------------------------------------------------

typedef __attribute__((ext_vector_type(16))) __bf16 v16bf;
typedef __attribute__((ext_vector_type(8)))  float  v8f;

union FragBF {
    v16bf v;
    uint4 q[2];
};

#define B_    2
#define T_    2048
#define E_    2048
#define H_    16
#define HD_   128
#define PAST_ 2048
#define TT_   4096  // total key length

// 1/sqrt(128) * log2(e): folded into Q so QK^T scores are in log2 domain.
#define QSCALE_LOG2E 0.12751743214862316f

// ---------------- fp32 -> bf16 cast (float4 at a time) ----------------
__global__ void cast_bf16_kernel(const float* __restrict__ src,
                                 __bf16* __restrict__ dst, int n4) {
    int i = blockIdx.x * blockDim.x + threadIdx.x;
    if (i >= n4) return;
    float4 v = ((const float4*)src)[i];
    __bf16 t[4] = {(__bf16)v.x, (__bf16)v.y, (__bf16)v.z, (__bf16)v.w};
    ((uint2*)dst)[i] = *(uint2*)t;
}

// -------- copy past cache [B*H,PAST,HD] into [B*H,TT,HD] (+opt bf16) -------
__global__ void copy_cache_kernel(const float* __restrict__ src,
                                  float* __restrict__ dstF,
                                  __bf16* __restrict__ dstB) {
    long long i = (long long)blockIdx.x * blockDim.x + threadIdx.x;
    const long long n4 = (long long)B_ * H_ * PAST_ * HD_ / 4;
    if (i >= n4) return;
    long long e = i * 4;
    long long plane = e / ((long long)PAST_ * HD_);
    long long inner = e - plane * ((long long)PAST_ * HD_);
    long long d = plane * (long long)TT_ * HD_ + inner;
    float4 v = ((const float4*)src)[i];
    *(float4*)(dstF + d) = v;
    if (dstB) {
        __bf16 t[4] = {(__bf16)v.x, (__bf16)v.y, (__bf16)v.z, (__bf16)v.w};
        *(uint2*)(dstB + d) = *(uint2*)t;
    }
}

// -------- V [B*H,TT,HD] fp32 -> Vt [B*H,HD,TT] bf16 (LDS tile transpose) ----
__global__ void transpose_v_kernel(const float* __restrict__ Vf,
                                   __bf16* __restrict__ Vt) {
    __shared__ float tile[32][33];
    int bh = blockIdx.z;
    int t0 = blockIdx.x * 32;
    int d0 = blockIdx.y * 32;
    const float* src = Vf + (long long)bh * TT_ * HD_;
    __bf16*      dst = Vt + (long long)bh * HD_ * TT_;
    int x = threadIdx.x;   // 0..31
    int y0 = threadIdx.y;  // 0..7
    #pragma unroll
    for (int i = 0; i < 4; ++i) {
        int y = y0 + i * 8;
        tile[y][x] = src[(long long)(t0 + y) * HD_ + d0 + x];
    }
    __syncthreads();
    #pragma unroll
    for (int i = 0; i < 4; ++i) {
        int y = y0 + i * 8;
        dst[(long long)(d0 + y) * TT_ + t0 + x] = (__bf16)tile[x][y];
    }
}

// ---------------------------------------------------------------------------
// GEMM: out = A[M,K] @ W[N,K]^T + bias.  bf16 inputs, f32 accum via WMMA.
// Block: 256 thr (8 waves, 2x4), block tile 64x256, wave tile 32x64.
// MODE 0: f32 row-major [M,N] -> outF
// MODE 1: bf16 -> outB laid out [B,H,T,hd], scaled by oscale (Q)
// MODE 2: f32 -> outF at [B,H,PAST+t,hd] (+ optional bf16 same layout)
// ---------------------------------------------------------------------------
template <int MODE>
__launch_bounds__(256)
__global__ void gemm_bf16_kernel(const __bf16* __restrict__ A,
                                 const __bf16* __restrict__ W,
                                 const float* __restrict__ bias,
                                 float* __restrict__ outF,
                                 __bf16* __restrict__ outB,
                                 float oscale,
                                 int M, int N, int K) {
    const int wave = threadIdx.x >> 5;
    const int lane = threadIdx.x & 31;
    const int lh = lane & 15;
    const int hi = lane >> 4;
    const int wm = wave & 1;   // 0..1
    const int wn = wave >> 1;  // 0..3
    const int m0 = blockIdx.x * 64 + wm * 32;
    const int n0 = blockIdx.y * 256 + wn * 64;

    v8f acc[2][4] = {};

    for (int k0 = 0; k0 < K; k0 += 32) {
        FragBF a[2], b[4];
        #pragma unroll
        for (int i = 0; i < 2; ++i) {
            const __bf16* p = A + (long long)(m0 + i * 16 + lh) * K + k0 + hi * 8;
            a[i].q[0] = *(const uint4*)p;
            a[i].q[1] = *(const uint4*)(p + 16);
        }
        #pragma unroll
        for (int j = 0; j < 4; ++j) {
            const __bf16* p = W + (long long)(n0 + j * 16 + lh) * K + k0 + hi * 8;
            b[j].q[0] = *(const uint4*)p;
            b[j].q[1] = *(const uint4*)(p + 16);
        }
        #pragma unroll
        for (int i = 0; i < 2; ++i) {
            #pragma unroll
            for (int j = 0; j < 4; ++j) {
                acc[i][j] = __builtin_amdgcn_wmma_f32_16x16x32_bf16(
                    false, a[i].v, false, b[j].v, (short)0, acc[i][j], false, false);
            }
        }
    }

    #pragma unroll
    for (int i = 0; i < 2; ++i) {
        #pragma unroll
        for (int j = 0; j < 4; ++j) {
            const int col = n0 + j * 16 + lh;
            const float bv = bias[col];
            #pragma unroll
            for (int r = 0; r < 8; ++r) {
                const int row = m0 + i * 16 + hi * 8 + r;
                float val = acc[i][j][r] + bv;
                if (MODE == 1) val *= oscale;
                if (MODE == 0) {
                    outF[(long long)row * N + col] = val;
                } else {
                    const int bb = row >> 11;        // / T_
                    const int tt = row & (T_ - 1);
                    const int hh = col >> 7;         // / HD_
                    const int dd = col & (HD_ - 1);
                    if (MODE == 1) {
                        long long idx = (((long long)bb * H_ + hh) * T_ + tt) * HD_ + dd;
                        outB[idx] = (__bf16)val;
                    } else {
                        long long idx =
                            (((long long)bb * H_ + hh) * TT_ + (PAST_ + tt)) * HD_ + dd;
                        outF[idx] = val;
                        if (outB) outB[idx] = (__bf16)val;
                    }
                }
            }
        }
    }
}

// ---------------------------------------------------------------------------
// One 32-key flash-attention chunk. Scores arrive already in log2 domain.
// Row sums of P come from P @ ones via one extra WMMA (C layout matches the
// l-accumulator layout exactly). MASKED only for the causal-edge chunks.
// ---------------------------------------------------------------------------
template <bool MASKED>
__device__ __forceinline__ void attn_chunk(
    int kbase, int q0, int lh, int hi,
    const __bf16* __restrict__ Kp, const __bf16* __restrict__ Vp,
    const FragBF (&qf)[4], const FragBF& ones, v8f (&o)[8],
    float (&mrow)[8], float (&lrow)[8], __bf16 (*psh)[32]) {

    // prefetch K rows two chunks ahead (one cacheline per lane), WGP scope
    __builtin_prefetch(Kp + (long long)(kbase + 64 + (hi * 16 + lh)) * HD_, 0, 3);

    v8f s0 = {}, s1 = {};
    #pragma unroll
    for (int c = 0; c < 4; ++c) {
        FragBF kb0, kb1;
        const __bf16* p0 = Kp + (long long)(kbase + lh) * HD_ + c * 32 + hi * 8;
        kb0.q[0] = *(const uint4*)p0;
        kb0.q[1] = *(const uint4*)(p0 + 16);
        const __bf16* p1 = p0 + 16 * HD_;
        kb1.q[0] = *(const uint4*)p1;
        kb1.q[1] = *(const uint4*)(p1 + 16);
        s0 = __builtin_amdgcn_wmma_f32_16x16x32_bf16(
            false, qf[c].v, false, kb0.v, (short)0, s0, false, false);
        s1 = __builtin_amdgcn_wmma_f32_16x16x32_bf16(
            false, qf[c].v, false, kb1.v, (short)0, s1, false, false);
    }

    float mnew[8], alpha[8];
    #pragma unroll
    for (int r = 0; r < 8; ++r) {
        float v0 = s0[r];
        float v1 = s1[r];
        if (MASKED) {
            const int qpos = PAST_ + q0 + hi * 8 + r;
            v0 = (kbase + lh <= qpos) ? v0 : -1.0e38f;
            v1 = (kbase + 16 + lh <= qpos) ? v1 : -1.0e38f;
            s0[r] = v0;
            s1[r] = v1;
        }
        float mx = fmaxf(v0, v1);
        mx = fmaxf(mx, __shfl_xor(mx, 1, 32));
        mx = fmaxf(mx, __shfl_xor(mx, 2, 32));
        mx = fmaxf(mx, __shfl_xor(mx, 4, 32));
        mx = fmaxf(mx, __shfl_xor(mx, 8, 32));
        mnew[r] = fmaxf(mrow[r], mx);
        alpha[r] = __builtin_amdgcn_exp2f(mrow[r] - mnew[r]);
        mrow[r] = mnew[r];
    }
    #pragma unroll
    for (int r = 0; r < 8; ++r) {
        float p0 = __builtin_amdgcn_exp2f(s0[r] - mnew[r]);
        float p1 = __builtin_amdgcn_exp2f(s1[r] - mnew[r]);
        psh[hi * 8 + r][lh] = (__bf16)p0;
        psh[hi * 8 + r][16 + lh] = (__bf16)p1;
    }
    #pragma unroll
    for (int t = 0; t < 8; ++t) {
        #pragma unroll
        for (int r = 0; r < 8; ++r) o[t][r] *= alpha[r];
    }
    FragBF pf;
    {
        const __bf16* base = &psh[lh][0];
        pf.q[0] = *(const uint4*)(base + hi * 8);
        pf.q[1] = *(const uint4*)(base + hi * 8 + 16);
    }
    // rowsum(P) via WMMA against ones: every column of C holds the row sum,
    // and the C layout puts row (hi*8+r) in element r of this lane.
    {
        v8f ls = {};
        ls = __builtin_amdgcn_wmma_f32_16x16x32_bf16(
            false, pf.v, false, ones.v, (short)0, ls, false, false);
        #pragma unroll
        for (int r = 0; r < 8; ++r) lrow[r] = lrow[r] * alpha[r] + ls[r];
    }
    #pragma unroll
    for (int t = 0; t < 8; ++t) {
        FragBF vb;
        const __bf16* p = Vp + (long long)(t * 16 + lh) * TT_ + kbase + hi * 8;
        vb.q[0] = *(const uint4*)p;
        vb.q[1] = *(const uint4*)(p + 16);
        o[t] = __builtin_amdgcn_wmma_f32_16x16x32_bf16(
            false, pf.v, false, vb.v, (short)0, o[t], false, false);
    }
}

// ---------------------------------------------------------------------------
// Flash attention. Block = 128 thr (4 waves); each wave owns 16 q rows.
// Unmasked main loop over fully-causal chunks, short masked tail.
// ---------------------------------------------------------------------------
__launch_bounds__(128)
__global__ void flash_attn_kernel(const __bf16* __restrict__ Qb,
                                  const __bf16* __restrict__ Kb,
                                  const __bf16* __restrict__ Vt,
                                  __bf16* __restrict__ attb) {
    __shared__ __bf16 pshare[4][16][32];
    const int wave = threadIdx.x >> 5;
    const int lane = threadIdx.x & 31;
    const int lh = lane & 15;
    const int hi = lane >> 4;
    const int b = blockIdx.z, h = blockIdx.y;
    const int q0 = blockIdx.x * 64 + wave * 16;

    const __bf16* Qp = Qb + (((long long)(b * H_ + h)) * T_ + q0) * HD_;
    const __bf16* Kp = Kb + ((long long)(b * H_ + h)) * TT_ * HD_;
    const __bf16* Vp = Vt + ((long long)(b * H_ + h)) * HD_ * TT_;

    FragBF qf[4];
    #pragma unroll
    for (int c = 0; c < 4; ++c) {
        const __bf16* p = Qp + (long long)lh * HD_ + c * 32 + hi * 8;
        qf[c].q[0] = *(const uint4*)p;
        qf[c].q[1] = *(const uint4*)(p + 16);
    }

    FragBF ones;
    #pragma unroll
    for (int e = 0; e < 16; ++e) ones.v[e] = (__bf16)1.0f;

    v8f o[8] = {};
    float mrow[8], lrow[8];
    #pragma unroll
    for (int r = 0; r < 8; ++r) { mrow[r] = -1.0e38f; lrow[r] = 0.f; }

    const int qmin = PAST_ + q0;            // smallest query position in wave
    const int nfull = (qmin + 1) >> 5;      // chunks valid for all 16 rows
    const int nch = (qmin + 15 + 32) >> 5;  // total chunks incl. causal edge

    for (int ch = 0; ch < nfull; ++ch)
        attn_chunk<false>(ch * 32, q0, lh, hi, Kp, Vp, qf, ones, o, mrow, lrow,
                          pshare[wave]);
    for (int ch = nfull; ch < nch; ++ch)
        attn_chunk<true>(ch * 32, q0, lh, hi, Kp, Vp, qf, ones, o, mrow, lrow,
                         pshare[wave]);

    float inv[8];
    #pragma unroll
    for (int r = 0; r < 8; ++r) inv[r] = 1.0f / lrow[r];
    #pragma unroll
    for (int t = 0; t < 8; ++t) {
        #pragma unroll
        for (int r = 0; r < 8; ++r) {
            const int row = q0 + hi * 8 + r;
            const int col = h * HD_ + t * 16 + lh;
            attb[((long long)b * T_ + row) * E_ + col] = (__bf16)(o[t][r] * inv[r]);
        }
    }
}

// ---------------------------------------------------------------------------
extern "C" void kernel_launch(void* const* d_in, const int* in_sizes, int n_in,
                              void* d_out, int out_size, void* d_ws, size_t ws_size,
                              hipStream_t stream) {
    (void)in_sizes; (void)n_in; (void)out_size; (void)ws_size;
    const float* x  = (const float*)d_in[0];
    const float* kc = (const float*)d_in[1];
    const float* vc = (const float*)d_in[2];
    const float* Wq = (const float*)d_in[3];
    const float* bq = (const float*)d_in[4];
    const float* Wk = (const float*)d_in[5];
    const float* bk = (const float*)d_in[6];
    const float* Wv = (const float*)d_in[7];
    const float* bv = (const float*)d_in[8];
    const float* Wo = (const float*)d_in[9];
    const float* bo = (const float*)d_in[10];
    float* out = (float*)d_out;

    char* ws = (char*)d_ws;
    __bf16* xb  = (__bf16*)(ws + 0ull);          // 16 MiB  x bf16
    __bf16* wqb = (__bf16*)(ws + 16777216ull);   //  8 MiB each weight
    __bf16* wkb = (__bf16*)(ws + 25165824ull);
    __bf16* wvb = (__bf16*)(ws + 33554432ull);
    __bf16* wob = (__bf16*)(ws + 41943040ull);
    __bf16* qb  = (__bf16*)(ws + 50331648ull);   // 16 MiB  Q  [B,H,T,hd]
    __bf16* kbf = (__bf16*)(ws + 67108864ull);   // 32 MiB  K  [B,H,TT,hd]
    __bf16* vt  = (__bf16*)(ws + 100663296ull);  // 32 MiB  V^T[B,H,hd,TT]
    __bf16* att = (__bf16*)(ws + 134217728ull);  // 16 MiB  att out [B,T,E]

    float* outO = out;
    float* outK = out + 8388608ull;   // K output region [B,H,TT,hd]
    float* outV = out + 25165824ull;  // V output region

    {
        int n4 = (B_ * T_ * E_) / 4;
        cast_bf16_kernel<<<(n4 + 255) / 256, 256, 0, stream>>>(x, xb, n4);
        int w4 = (E_ * E_) / 4;
        cast_bf16_kernel<<<(w4 + 255) / 256, 256, 0, stream>>>(Wq, wqb, w4);
        cast_bf16_kernel<<<(w4 + 255) / 256, 256, 0, stream>>>(Wk, wkb, w4);
        cast_bf16_kernel<<<(w4 + 255) / 256, 256, 0, stream>>>(Wv, wvb, w4);
        cast_bf16_kernel<<<(w4 + 255) / 256, 256, 0, stream>>>(Wo, wob, w4);
    }

    dim3 gg(B_ * T_ / 64, E_ / 256);  // (64, 8)
    gemm_bf16_kernel<1><<<gg, 256, 0, stream>>>(xb, wqb, bq, nullptr, qb,
                                                QSCALE_LOG2E, B_ * T_, E_, E_);
    gemm_bf16_kernel<2><<<gg, 256, 0, stream>>>(xb, wkb, bk, outK, kbf,
                                                1.0f, B_ * T_, E_, E_);
    gemm_bf16_kernel<2><<<gg, 256, 0, stream>>>(xb, wvb, bv, outV, nullptr,
                                                1.0f, B_ * T_, E_, E_);
    {
        long long n4 = (long long)B_ * H_ * PAST_ * HD_ / 4;
        int blocks = (int)((n4 + 255) / 256);
        copy_cache_kernel<<<blocks, 256, 0, stream>>>(kc, outK, kbf);
        copy_cache_kernel<<<blocks, 256, 0, stream>>>(vc, outV, nullptr);
    }
    transpose_v_kernel<<<dim3(TT_ / 32, HD_ / 32, B_ * H_), dim3(32, 8), 0,
                         stream>>>(outV, vt);
    flash_attn_kernel<<<dim3(T_ / 64, H_, B_), 128, 0, stream>>>(qb, kbf, vt, att);
    gemm_bf16_kernel<0><<<gg, 256, 0, stream>>>(att, wob, bo, outO, nullptr,
                                                1.0f, B_ * T_, E_, E_);
}
